// VectorQuantizer_23184233464491
// MI455X (gfx1250) — matrix operand order repaired and checked by reading, not compiled, not run on hardware
//
#include <hip/hip_runtime.h>
#include <hip/hip_bf16.h>
#include <math.h>

typedef __attribute__((ext_vector_type(16))) __bf16 v16bf;
typedef __attribute__((ext_vector_type(8)))  __bf16 v8bf;
typedef __attribute__((ext_vector_type(4)))  __bf16 v4bf;
typedef __attribute__((ext_vector_type(8)))  float  v8f;

// gcc-style vector + address-space-qualified pointers for the async builtin
typedef int v4i __attribute__((vector_size(16)));
typedef __attribute__((address_space(1))) v4i* gv4i_p;   // global (AS1)
typedef __attribute__((address_space(3))) v4i* lv4i_p;   // LDS (AS3)

#define D_DIM     256
#define K_CB      1024
#define HW        1024          // H*W
#define BSTRIDE   262144        // D*H*W floats per batch image
#define QELEMS    16777216      // 64*256*32*32
#define N_ROWS    65536         // B*H*W

#define LDS_STRIDE 264          // padded row stride in halfs (bank-conflict free)

// ---- CDNA5 async global->LDS DMA (ASYNCcnt-tracked), with fallback --------
#if defined(__has_builtin)
#if __has_builtin(__builtin_amdgcn_global_load_async_to_lds_b128)
#define HAVE_ASYNC_LDS 1
#endif
#endif
#ifndef HAVE_ASYNC_LDS
#define HAVE_ASYNC_LDS 0
#endif

#if HAVE_ASYNC_LDS
#if __has_builtin(__builtin_amdgcn_s_wait_asynccnt)
#define WAIT_ASYNC(N) __builtin_amdgcn_s_wait_asynccnt(N)
#else
#define WAIT_ASYNC(N) asm volatile("s_wait_asynccnt %0" ::"n"(N) : "memory")
#endif
#endif

// ---------------------------------------------------------------------------
// Kernel 1a: codebook norms |e_k|^2, zero histogram + loss accumulator
// ---------------------------------------------------------------------------
__global__ __launch_bounds__(256) void vq_init_kernel(
    const float* __restrict__ codebook, float* __restrict__ norms,
    unsigned* __restrict__ counts, float* __restrict__ lossAcc)
{
    const int k = blockIdx.x * 256 + threadIdx.x;   // 1024 total
    const float* row = codebook + (size_t)k * D_DIM;
    float s = 0.0f;
    for (int d = 0; d < D_DIM; ++d) s += row[d] * row[d];
    norms[k]  = s;
    counts[k] = 0u;
    if (k == 0) *lossAcc = 0.0f;
}

// ---------------------------------------------------------------------------
// Kernel 1b: one-time fp32 -> split-bf16 (hi + residual lo) codebook convert.
// 1 MB total, L2-resident (192 MB); removes 64x redundant per-WG conversion.
// ---------------------------------------------------------------------------
__global__ __launch_bounds__(256) void vq_convert_kernel(
    const float* __restrict__ codebook, __bf16* __restrict__ cbhi,
    __bf16* __restrict__ cblo)
{
    const int t = blockIdx.x * 256 + threadIdx.x;   // 65536 threads, 4 floats each
    const float4 x = *(const float4*)(codebook + (size_t)t * 4);
    v4bf h, l;
    const float xs[4] = {x.x, x.y, x.z, x.w};
#pragma unroll
    for (int i = 0; i < 4; ++i) {
        const __bf16 hi = (__bf16)xs[i];
        h[i] = hi;
        l[i] = (__bf16)(xs[i] - (float)hi);
    }
    *(v4bf*)(cbhi + (size_t)t * 4) = h;
    *(v4bf*)(cblo + (size_t)t * 4) = l;
}

// ---------------------------------------------------------------------------
// Stage one 16x256 bf16 tile (8 KB) from global (ws) into padded LDS.
// 2 x b128 per thread; async DMA on CDNA5, synchronous copy as fallback.
// ---------------------------------------------------------------------------
__device__ __forceinline__ void stage_tile(const __bf16* __restrict__ src,
                                           __bf16* dst, int kt, int tid)
{
#pragma unroll
    for (int j = 0; j < 2; ++j) {
        const int q   = j * 256 + tid;       // 512 16B-chunks per tile
        const int row = q >> 5;              // 32 chunks per 512B row
        const int cb  = q & 31;
        const int gof = (kt * 16 + row) * D_DIM + cb * 8;   // bf16 elems
        const int lof = row * LDS_STRIDE + cb * 8;
#if HAVE_ASYNC_LDS
        __builtin_amdgcn_global_load_async_to_lds_b128(
            (gv4i_p)(src + gof), (lv4i_p)(dst + lof), 0, 0);
#else
        *(uint4*)(dst + lof) = *(const uint4*)(src + gof);
#endif
    }
}

// ---------------------------------------------------------------------------
// Kernel 2: fused distance-GEMM + argmin, split-bf16 WMMA, double-buffered
// async LDS pipeline. Each wave32 owns a 16-row tile; WG = 8 waves = 128 rows.
// ---------------------------------------------------------------------------
__global__ __launch_bounds__(256) void vq_argmin_kernel(
    const float* __restrict__ latents, const __bf16* __restrict__ cbhi,
    const __bf16* __restrict__ cblo, const float* __restrict__ norms,
    int* __restrict__ idx_out, float* __restrict__ idxf_out,
    unsigned* __restrict__ counts)
{
    __shared__ __align__(16) __bf16 BhiD[2][16 * LDS_STRIDE];
    __shared__ __align__(16) __bf16 BloD[2][16 * LDS_STRIDE];

    const int tid  = threadIdx.x;
    const int lane = tid & 31;
    const int wave = tid >> 5;
    const int m    = lane & 15;     // A-row within tile; also B/C column
    const int half = lane >> 4;
    const int ko   = half * 8;      // A-fragment K offset per ISA layout

    const int rowBase = blockIdx.x * 128 + wave * 16;
    const int n = rowBase + m;
    const int b = n >> 10;
    const int p = n & (HW - 1);
    const float* lat = latents + (size_t)b * BSTRIDE + p;  // element d at lat[d*HW]

    // ---- Load the 16x256 fp32 A tile once, split into bf16 hi/lo fragments.
    // A 16x32 fragment in this code's contiguous k order (B matches): lane(0-15)=row M, element e -> K = half*16 + e
    v16bf ahi[8], alo[8];
#pragma unroll
    for (int c = 0; c < 8; ++c) {
#pragma unroll
        for (int e = 0; e < 16; ++e) {
            const int d  = 32 * c + half * 16 + e;
            const float x = lat[(size_t)d * HW];
            const __bf16 h = (__bf16)x;
            const __bf16 l = (__bf16)(x - (float)h);
            ahi[c][e] = h;
            alo[c][e] = l;
        }
    }

    float minv[8];
    int   mini[8];
#pragma unroll
    for (int r = 0; r < 8; ++r) { minv[r] = 3.4e38f; mini[r] = 0; }

    // Prologue: start DMA of tile 0 into buffer 0.
    stage_tile(cbhi, &BhiD[0][0], 0, tid);
    stage_tile(cblo, &BloD[0][0], 0, tid);

    for (int kt = 0; kt < 64; ++kt) {
        const int cur = kt & 1;
        // Kick off DMA for tile kt+1 into the other buffer (its previous
        // readers finished at the end-of-iteration barrier of kt-1).
        if (kt + 1 < 64) {
            const int nxt = (kt + 1) & 1;
            stage_tile(cbhi, &BhiD[nxt][0], kt + 1, tid);
            stage_tile(cblo, &BloD[nxt][0], kt + 1, tid);
        }
#if HAVE_ASYNC_LDS
        // Async loads complete in order: waiting <=4 leaves only the 4 loads
        // of tile kt+1 outstanding => tile kt is resident for this wave.
        if (kt + 1 < 64) { WAIT_ASYNC(4); } else { WAIT_ASYNC(0); }
#endif
        __syncthreads();   // all waves' portions of tile kt resident

        v8f acc = {};
#pragma unroll
        for (int c = 0; c < 8; ++c) {
            // B 32x16 fragment, same contiguous k order as A: lane col = m, element e -> K = half*16 + e
            const __bf16* bp = &BhiD[cur][m * LDS_STRIDE + 32 * c + half * 16];
            const __bf16* lp = &BloD[cur][m * LDS_STRIDE + 32 * c + half * 16];
            v8bf h0 = *(const v8bf*)(bp);
            v8bf h1 = *(const v8bf*)(bp + 8);
            v8bf l0 = *(const v8bf*)(lp);
            v8bf l1 = *(const v8bf*)(lp + 8);
            v16bf bh = __builtin_shufflevector(h0, h1, 0,1,2,3,4,5,6,7,8,9,10,11,12,13,14,15);
            v16bf bl = __builtin_shufflevector(l0, l1, 0,1,2,3,4,5,6,7,8,9,10,11,12,13,14,15);
            // split-bf16: x.e ~= xh.eh + xh.el + xl.eh  (xl.el ~ 2^-32, dropped)
            acc = __builtin_amdgcn_wmma_f32_16x16x32_bf16(false, ahi[c], false, bh,
                                                          (short)0, acc, false, false);
            acc = __builtin_amdgcn_wmma_f32_16x16x32_bf16(false, ahi[c], false, bl,
                                                          (short)0, acc, false, false);
            acc = __builtin_amdgcn_wmma_f32_16x16x32_bf16(false, alo[c], false, bh,
                                                          (short)0, acc, false, false);
        }

        // C layout: lane -> col = m, rows M = half*8 + r (r = vgpr index)
        const float nrm = norms[kt * 16 + m];
        const int   gk  = kt * 16 + m;
#pragma unroll
        for (int r = 0; r < 8; ++r) {
            const float dist = nrm - 2.0f * acc[r];
            if (dist < minv[r]) { minv[r] = dist; mini[r] = gk; }
        }
        __syncthreads();   // all waves done reading tile kt before overwrite
    }

    // Cross-lane min within each 16-lane group (xor masks < 16 stay in-group).
#pragma unroll
    for (int r = 0; r < 8; ++r) {
        float v = minv[r];
        int   i = mini[r];
#pragma unroll
        for (int s = 1; s < 16; s <<= 1) {
            const float ov = __shfl_xor(v, s, 32);
            const int   oi = __shfl_xor(i, s, 32);
            if (ov < v || (ov == v && oi < i)) { v = ov; i = oi; }
        }
        if (m == 0) {                       // lanes 0 and 16 write 8 rows each
            const int row = rowBase + half * 8 + r;
            idx_out[row]  = i;
            idxf_out[row] = (float)i;
            atomicAdd(&counts[i], 1u);
        }
    }
}

// ---------------------------------------------------------------------------
// Kernel 3: gather quantized vectors (coalesced float4 stores) + fused
// commitment-loss reduction (global_atomic_add_f32).
// ---------------------------------------------------------------------------
__global__ __launch_bounds__(256) void vq_gather_loss_kernel(
    const float* __restrict__ latents, const float* __restrict__ codebook,
    const int* __restrict__ idx, float* __restrict__ quant,
    float* __restrict__ lossAcc)
{
    const long long t = (long long)blockIdx.x * 256 + threadIdx.x; // 4194304 total
    const int pg = (int)(t & 255);
    const int d  = (int)((t >> 8) & 255);
    const int b  = (int)(t >> 16);
    const int p  = pg * 4;

    const size_t obase = (size_t)b * BSTRIDE + (size_t)d * HW + p;
    const int* ib = idx + b * HW + p;

    float4 q;
    q.x = codebook[(size_t)ib[0] * D_DIM + d];
    q.y = codebook[(size_t)ib[1] * D_DIM + d];
    q.z = codebook[(size_t)ib[2] * D_DIM + d];
    q.w = codebook[(size_t)ib[3] * D_DIM + d];

    const float4 x = *(const float4*)(latents + obase);
    *(float4*)(quant + obase) = q;

    const float dx = x.x - q.x, dy = x.y - q.y, dz = x.z - q.z, dw = x.w - q.w;
    float s = dx * dx + dy * dy + dz * dz + dw * dw;

    __shared__ float red[256];
    red[threadIdx.x] = s;
    __syncthreads();
    for (int st = 128; st > 0; st >>= 1) {
        if (threadIdx.x < st) red[threadIdx.x] += red[threadIdx.x + st];
        __syncthreads();
    }
    if (threadIdx.x == 0) atomicAdd(lossAcc, red[0]);
}

// ---------------------------------------------------------------------------
// Kernel 4: vq_loss + perplexity scalars
// ---------------------------------------------------------------------------
__global__ __launch_bounds__(1024) void vq_finalize_kernel(
    const unsigned* __restrict__ counts, const float* __restrict__ lossAcc,
    float* __restrict__ scalars)
{
    __shared__ float red[1024];
    const int t = threadIdx.x;
    const float pr = (float)counts[t] * (1.0f / (float)N_ROWS);
    red[t] = pr * logf(pr + 1e-10f);
    __syncthreads();
    for (int s = 512; s > 0; s >>= 1) {
        if (t < s) red[t] += red[t + s];
        __syncthreads();
    }
    if (t == 0) {
        scalars[0] = 0.25f * lossAcc[0] * (1.0f / (float)QELEMS);  // vq_loss
        scalars[1] = expf(-red[0]);                                // perplexity
    }
}

// ---------------------------------------------------------------------------
extern "C" void kernel_launch(void* const* d_in, const int* in_sizes, int n_in,
                              void* d_out, int out_size, void* d_ws, size_t ws_size,
                              hipStream_t stream)
{
    const float* latents  = (const float*)d_in[0];   // (64,256,32,32) fp32
    const float* codebook = (const float*)d_in[1];   // (1024,256) fp32

    float* out     = (float*)d_out;
    float* quant   = out;                   // 16777216 elems
    float* scalars = out + QELEMS;          // vq_loss, perplexity
    float* idxf    = out + QELEMS + 2;      // 65536 indices (as float)

    char* ws = (char*)d_ws;
    float*    norms   = (float*)ws;                           // 1024 f32
    unsigned* counts  = (unsigned*)(ws + 4096);               // 1024 u32
    int*      idx     = (int*)(ws + 8192);                    // 65536 i32
    float*    lossAcc = (float*)(ws + 270336);                // 1 f32
    __bf16*   cbhi    = (__bf16*)(ws + 270848);               // 1024x256 bf16
    __bf16*   cblo    = (__bf16*)(ws + 795136);               // 1024x256 bf16

    vq_init_kernel<<<4, 256, 0, stream>>>(codebook, norms, counts, lossAcc);
    vq_convert_kernel<<<256, 256, 0, stream>>>(codebook, cbhi, cblo);
    vq_argmin_kernel<<<512, 256, 0, stream>>>(latents, cbhi, cblo, norms,
                                              idx, idxf, counts);
    vq_gather_loss_kernel<<<QELEMS / (4 * 256), 256, 0, stream>>>(
        latents, codebook, idx, quant, lossAcc);
    vq_finalize_kernel<<<1, 1024, 0, stream>>>(counts, lossAcc, scalars);
}
